// TestModel_23192823399121
// MI455X (gfx1250) — matrix operand order repaired and checked by reading, not compiled, not run on hardware
//
#include <hip/hip_runtime.h>
#include <math.h>
#include <stdint.h>

// ---------------- CDNA5 WMMA plumbing (wave32, 16x16x32 bf16 -> f32) -------
typedef __bf16 bf16;
typedef __attribute__((ext_vector_type(16))) bf16  v16bf;
typedef __attribute__((ext_vector_type(8)))  float v8f;

__device__ __forceinline__ v8f wmma_bf16(v16bf a, v16bf b, v8f c) {
  return __builtin_amdgcn_wmma_f32_16x16x32_bf16(false, a, false, b, (short)0, c,
                                                 false, false);
}

// A fragment: 16x32 tile, row-major f32 source (converted to bf16 on the fly).
// Lane l (<16): row=l, K elems {half*8+j, 16+half*8+j}; lanes 16..31: half=1.
__device__ __forceinline__ v16bf a_frag_f32(const float* __restrict__ A, int ld,
                                            int kbase) {
  int l = threadIdx.x & 31;
  const float* p = A + (size_t)(l & 15) * ld + kbase + (l >> 4) * 8;
  v16bf r;
#pragma unroll
  for (int j = 0; j < 8; ++j) { r[j] = (bf16)p[j]; r[8 + j] = (bf16)p[16 + j]; }
  return r;
}

__device__ __forceinline__ v16bf a_frag_bf(const bf16* A, int ld, int kbase) {
  int l = threadIdx.x & 31;
  const bf16* p = A + (size_t)(l & 15) * ld + kbase + (l >> 4) * 8;
  v16bf r;
#pragma unroll
  for (int j = 0; j < 8; ++j) { r[j] = p[j]; r[8 + j] = p[16 + j]; }
  return r;
}

// B fragment from pre-packed fragment-major storage: per (frag, lane) the 16
// bf16 elements are contiguous (32B, 32B-aligned) -> 2x global_load_b128.
__device__ __forceinline__ v16bf load_bfrag(const bf16* __restrict__ base,
                                            int frag) {
  return *(const v16bf*)(base + ((size_t)frag * 32 + (threadIdx.x & 31)) * 16);
}

// C/D element (reg r, lane l) maps to row = r + 8*(l>>4), col = l&15.
#define LRELU(x) ((x) > 0.0f ? (x) : 0.01f * (x))

// ---------------- deterministic stand-in for jax noise ---------------------
__device__ __forceinline__ uint32_t pcg_hash(uint32_t x) {
  uint32_t s = x * 747796405u + 2891336453u;
  uint32_t w = ((s >> ((s >> 28u) + 4u)) ^ s) * 277803737u;
  return (w >> 22u) ^ w;
}
__device__ __forceinline__ float hash_normal(uint32_t s) {
  uint32_t h1 = pcg_hash(s), h2 = pcg_hash(s ^ 0x9E3779B9u);
  float u1 = ((h1 >> 8) + 1u) * (1.0f / 16777216.0f);
  float u2 = (h2 >> 8) * (1.0f / 16777216.0f);
  return sqrtf(fmaxf(-2.0f * __logf(u1), 0.0f)) * __cosf(6.28318530718f * u2);
}

// ---------------- prep: pack B matrices into WMMA fragment layout ----------
// Layout: dst[((c*nkc + kc)*32 + lane)*16 + j] =
//           B[kc*32 + (lane>>4)*8 + (j&7) + (j>>3)*16][c*16 + (lane&15)]
// where B is [Kdim x Npad]; source is row-major [Kdim x Nreal] (cols >= Nreal
// are zero-padded); transpose!=0 reads src as [Nreal x Kdim] (i.e. B = src^T).
__global__ void pack_b_kernel(const float* __restrict__ src, bf16* __restrict__ dst,
                              int Kdim, int Npad, int Nreal, int transpose) {
  int i = blockIdx.x * blockDim.x + threadIdx.x;
  int total = Kdim * Npad;
  if (i >= total) return;
  int j = i & 15;
  int l = (i >> 4) & 31;
  int rest = i >> 9;
  int nkc = Kdim >> 5;
  int kc = rest % nkc;
  int c = rest / nkc;
  int k = kc * 32 + (l >> 4) * 8 + (j & 7) + (j >> 3) * 16;
  int ncol = c * 16 + (l & 15);
  float v = 0.0f;
  if (ncol < Nreal)
    v = transpose ? src[(size_t)ncol * Kdim + k] : src[(size_t)k * Nreal + ncol];
  dst[i] = (bf16)v;
}

__global__ void zero4_kernel(float4* __restrict__ p, long long count4) {
  long long i = (long long)blockIdx.x * blockDim.x + threadIdx.x;
  long long stride = (long long)gridDim.x * blockDim.x;
  float4 z; z.x = z.y = z.z = z.w = 0.0f;
  for (; i < count4; i += stride) p[i] = z;
}

// ---------------- encoder: Z = l2n(lrelu(lrelu(X W1 + b1) W2 + b2)) --------
// one wave per 16-row tile; block = 8 waves
__global__ __launch_bounds__(256) void encoder_kernel(
    const float* __restrict__ X, const bf16* __restrict__ W1p,
    const float* __restrict__ b1, const bf16* __restrict__ W2p,
    const float* __restrict__ b2, float* __restrict__ Z, int ntiles) {
  __shared__ bf16 lds_t[8][16 * 128];  // 32 KB staged hidden activations
  int wave = threadIdx.x >> 5;
  int tile = blockIdx.x * 8 + wave;
  if (tile >= ntiles) return;
  int l = threadIdx.x & 31, col16 = l & 15, grp = l >> 4;
  const float* Arows = X + (size_t)tile * 16 * 256;

  // GEMM1: [16x256] x [256x128], W1p fragment index = c*8 + kc
  v8f acc[8] = {};
#pragma unroll
  for (int kc = 0; kc < 8; ++kc) {
    v16bf a = a_frag_f32(Arows, 256, kc * 32);
#pragma unroll
    for (int c = 0; c < 8; ++c)
      acc[c] = wmma_bf16(a, load_bfrag(W1p, c * 8 + kc), acc[c]);
  }
  bf16* t = lds_t[wave];
#pragma unroll
  for (int c = 0; c < 8; ++c) {
    int col = c * 16 + col16;
    float bias = b1[col];
#pragma unroll
    for (int r = 0; r < 8; ++r) {
      float v = acc[c][r] + bias;
      t[(r + 8 * grp) * 128 + col] = (bf16)LRELU(v);
    }
  }
  // GEMM2: [16x128] x [128x128], W2p fragment index = c*4 + kc
  v8f acc2[8] = {};
#pragma unroll
  for (int kc = 0; kc < 4; ++kc) {
    v16bf a = a_frag_bf(t, 128, kc * 32);
#pragma unroll
    for (int c = 0; c < 8; ++c)
      acc2[c] = wmma_bf16(a, load_bfrag(W2p, c * 4 + kc), acc2[c]);
  }
  // bias + lrelu + row L2 norm (shfl reduction within each 16-lane half)
  float ss[8] = {};
#pragma unroll
  for (int c = 0; c < 8; ++c) {
    float bias = b2[c * 16 + col16];
#pragma unroll
    for (int r = 0; r < 8; ++r) {
      float v = acc2[c][r] + bias;
      v = LRELU(v);
      acc2[c][r] = v;
      ss[r] += v * v;
    }
  }
#pragma unroll
  for (int r = 0; r < 8; ++r)
#pragma unroll
    for (int off = 8; off >= 1; off >>= 1) ss[r] += __shfl_xor(ss[r], off, 16);
#pragma unroll
  for (int r = 0; r < 8; ++r) {
    float inv = 1.0f / fmaxf(sqrtf(ss[r]), 1e-12f);
    size_t row = (size_t)tile * 16 + r + 8 * grp;
#pragma unroll
    for (int c = 0; c < 8; ++c)
      Z[row * 128 + c * 16 + col16] = acc2[c][r] * inv;
  }
}

// ---------------- fuse_Z = l2n((Z0+Z1)/2) ----------------------------------
__global__ void fuse_kernel(const float* __restrict__ Z0,
                            const float* __restrict__ Z1,
                            float* __restrict__ fuse) {
  __shared__ float red[128];
  size_t row = blockIdx.x;
  int c = threadIdx.x;
  float v = 0.5f * (Z0[row * 128 + c] + Z1[row * 128 + c]);
  red[c] = v * v;
  __syncthreads();
  for (int s = 64; s >= 1; s >>= 1) {
    if (c < s) red[c] += red[c + s];
    __syncthreads();
  }
  float inv = 1.0f / fmaxf(sqrtf(red[0]), 1e-12f);
  fuse[row * 128 + c] = v * inv;
}

// ---------------- anchor = l2n(a0 + prelu(a0 Mu + mb) + relu(a0 Zi + zb)*noise)
__global__ __launch_bounds__(256) void anchor_kernel(
    const float* __restrict__ fuse, const bf16* __restrict__ MuWp,
    const float* __restrict__ mu_b, const bf16* __restrict__ ZiWp,
    const float* __restrict__ zi_b, const float* __restrict__ prelu_w,
    float* __restrict__ anchor) {
  int tile = blockIdx.x * 8 + (threadIdx.x >> 5);  // 64 tiles of 16 rows
  int l = threadIdx.x & 31, col16 = l & 15, grp = l >> 4;
  const float* Arows = fuse + (size_t)tile * 16 * 128;
  v16bf afr[4];
#pragma unroll
  for (int kc = 0; kc < 4; ++kc) afr[kc] = a_frag_f32(Arows, 128, kc * 32);
  v8f mu[8] = {}, zi[8] = {};
#pragma unroll
  for (int kc = 0; kc < 4; ++kc)
#pragma unroll
    for (int c = 0; c < 8; ++c) {
      mu[c] = wmma_bf16(afr[kc], load_bfrag(MuWp, c * 4 + kc), mu[c]);
      zi[c] = wmma_bf16(afr[kc], load_bfrag(ZiWp, c * 4 + kc), zi[c]);
    }
  float ss[8] = {};
#pragma unroll
  for (int c = 0; c < 8; ++c) {
    int col = c * 16 + col16;
    float mb = mu_b[col], zb = zi_b[col], pw = prelu_w[col];
#pragma unroll
    for (int r = 0; r < 8; ++r) {
      int row = tile * 16 + r + 8 * grp;
      float a0 = Arows[(r + 8 * grp) * 128 + col];
      float m = mu[c][r] + mb; m = (m >= 0.0f) ? m : pw * m;           // PReLU
      float z = zi[c][r] + zb; z = (z > 0.0f) ? z : 0.0f;              // ReLU
      float v = a0 + m + z * hash_normal((uint32_t)(row * 128 + col));
      mu[c][r] = v;
      ss[r] += v * v;
    }
  }
#pragma unroll
  for (int r = 0; r < 8; ++r)
#pragma unroll
    for (int off = 8; off >= 1; off >>= 1) ss[r] += __shfl_xor(ss[r], off, 16);
#pragma unroll
  for (int r = 0; r < 8; ++r) {
    float inv = 1.0f / fmaxf(sqrtf(ss[r]), 1e-12f);
    size_t row = (size_t)tile * 16 + r + 8 * grp;
#pragma unroll
    for (int c = 0; c < 8; ++c)
      anchor[row * 128 + c * 16 + col16] = mu[c][r] * inv;
  }
}

// ---------------- S = Z anchor^T, per-row top-10 + softmax -> dense A ------
// one wave per 16-row tile; 16x1024 score tile staged in 64 KB LDS.
// anchorPk is anchor^T pre-packed into B-fragment layout: frag = nt*4 + kc.
__global__ __launch_bounds__(32) void score_topk_kernel(
    const float* __restrict__ Z, const bf16* __restrict__ anchorPk,
    float* __restrict__ Adense, int* __restrict__ top_idx,
    float* __restrict__ top_w) {
  __shared__ float sc[16 * 1024];
  int tile = blockIdx.x;
  int l = threadIdx.x, col16 = l & 15, grp = l >> 4;
  const float* Arows = Z + (size_t)tile * 16 * 128;
  v16bf afr[4];
#pragma unroll
  for (int kc = 0; kc < 4; ++kc) afr[kc] = a_frag_f32(Arows, 128, kc * 32);
  for (int nt = 0; nt < 64; ++nt) {
    if (nt + 1 < 64)  // gfx1250 global_prefetch_b8 on next packed B block
      __builtin_prefetch(anchorPk + ((size_t)(nt + 1) * 4 * 32) * 16, 0, 1);
    v8f acc = {};
#pragma unroll
    for (int kc = 0; kc < 4; ++kc)
      acc = wmma_bf16(afr[kc], load_bfrag(anchorPk, nt * 4 + kc), acc);
#pragma unroll
    for (int r = 0; r < 8; ++r)
      sc[(r + 8 * grp) * 1024 + nt * 16 + col16] = acc[r];
  }
  __syncthreads();
  if (l < 16) {
    size_t row = (size_t)tile * 16 + l;
    const float* srow = sc + l * 1024;
    float vals[10]; int idxs[10];
#pragma unroll
    for (int j = 0; j < 10; ++j) { vals[j] = -1e30f; idxs[j] = 0; }
    for (int mcol = 0; mcol < 1024; ++mcol) {
      float v = srow[mcol];
      if (v > vals[9]) {  // insertion into sorted top-10 (stable on ties)
        int p = 9;
        while (p > 0 && vals[p - 1] < v) { vals[p] = vals[p - 1]; idxs[p] = idxs[p - 1]; --p; }
        vals[p] = v; idxs[p] = mcol;
      }
    }
    float mx = vals[0], se = 0.0f, e[10];
#pragma unroll
    for (int j = 0; j < 10; ++j) { e[j] = __expf(vals[j] - mx); se += e[j]; }
    float inv = 1.0f / se;
#pragma unroll
    for (int j = 0; j < 10; ++j) {
      float w = e[j] * inv;
      Adense[row * 1024 + idxs[j]] = w;
      top_idx[row * 10 + j] = idxs[j];
      top_w[row * 10 + j] = w;
    }
  }
}

// ---------------- sparse anchor-graph conv pieces --------------------------
__global__ void colsum_kernel(const int* __restrict__ idx,
                              const float* __restrict__ w, float* __restrict__ d,
                              int total) {
  int i = blockIdx.x * blockDim.x + threadIdx.x;
  if (i < total) atomicAdd(&d[idx[i]], w[i]);
}

__global__ void dinv_kernel(const float* __restrict__ d, float* __restrict__ dinv) {
  int j = blockIdx.x * blockDim.x + threadIdx.x;
  if (j < 1024) dinv[j] = (d[j] > 0.0f) ? 1.0f / d[j] : 0.0f;
}

// t2 += A^T (A F): per sample compute af = sum_j w_j F[idx_j], scatter w_j*af
__global__ void conv_scatter_kernel(const int* __restrict__ idx,
                                    const float* __restrict__ w,
                                    const float* __restrict__ Fm,
                                    float* __restrict__ t2, int f) {
  size_t s = blockIdx.x;
  int c = threadIdx.x;  // blockDim == f
  int li[10]; float lw[10];
  float af = 0.0f;
#pragma unroll
  for (int j = 0; j < 10; ++j) {
    li[j] = idx[s * 10 + j];
    lw[j] = w[s * 10 + j];
    af += lw[j] * Fm[(size_t)li[j] * f + c];
  }
#pragma unroll
  for (int j = 0; j < 10; ++j) atomicAdd(&t2[(size_t)li[j] * f + c], lw[j] * af);
}

__global__ void scale_kernel(const float* __restrict__ t2,
                             const float* __restrict__ dinv,
                             float* __restrict__ outp, int f, int total) {
  int i = blockIdx.x * blockDim.x + threadIdx.x;
  if (i < total) outp[i] = t2[i] * dinv[i / f];
}

// ---------------- q = lrelu(conv1 @ W) : [1024x128]x[128x32] ---------------
__global__ __launch_bounds__(256) void agcn1_kernel(const float* __restrict__ conv1,
                                                    const bf16* __restrict__ Wp,
                                                    float* __restrict__ q) {
  int tile = blockIdx.x * 8 + (threadIdx.x >> 5);  // 64 tiles
  int l = threadIdx.x & 31, col16 = l & 15, grp = l >> 4;
  const float* Arows = conv1 + (size_t)tile * 16 * 128;
  v8f acc[2] = {};
#pragma unroll
  for (int kc = 0; kc < 4; ++kc) {
    v16bf a = a_frag_f32(Arows, 128, kc * 32);
#pragma unroll
    for (int c = 0; c < 2; ++c)
      acc[c] = wmma_bf16(a, load_bfrag(Wp, c * 4 + kc), acc[c]);
  }
#pragma unroll
  for (int c = 0; c < 2; ++c)
#pragma unroll
    for (int r = 0; r < 8; ++r) {
      float v = acc[c][r];
      q[(size_t)(tile * 16 + r + 8 * grp) * 32 + c * 16 + col16] = LRELU(v);
    }
}

// ---------------- Q = softmax(conv2 @ W2pad)[:, :10] -----------------------
__global__ __launch_bounds__(256) void agcn2_kernel(const float* __restrict__ conv2,
                                                    const bf16* __restrict__ Wp,
                                                    float* __restrict__ Q) {
  __shared__ float srow[8][16 * 16];
  int wave = threadIdx.x >> 5;
  int tile = blockIdx.x * 8 + wave;  // 64 tiles
  int l = threadIdx.x & 31, col16 = l & 15, grp = l >> 4;
  v16bf a = a_frag_f32(conv2 + (size_t)tile * 16 * 32, 32, 0);
  v16bf b = load_bfrag(Wp, 0);
  v8f acc = {};
  acc = wmma_bf16(a, b, acc);
#pragma unroll
  for (int r = 0; r < 8; ++r) srow[wave][(r + 8 * grp) * 16 + col16] = acc[r];
  __syncthreads();
  if (l < 16) {
    const float* sr = srow[wave] + l * 16;
    float mx = -1e30f;
#pragma unroll
    for (int j = 0; j < 10; ++j) mx = fmaxf(mx, sr[j]);
    float se = 0.0f, e[10];
#pragma unroll
    for (int j = 0; j < 10; ++j) { e[j] = __expf(sr[j] - mx); se += e[j]; }
    float inv = 1.0f / se;
#pragma unroll
    for (int j = 0; j < 10; ++j)
      Q[(size_t)(tile * 16 + l) * 10 + j] = e[j] * inv;
  }
}

// ---------------- host orchestration ---------------------------------------
extern "C" void kernel_launch(void* const* d_in, const int* in_sizes, int n_in,
                              void* d_out, int out_size, void* d_ws, size_t ws_size,
                              hipStream_t stream) {
  const float* X0      = (const float*)d_in[0];
  const float* X1      = (const float*)d_in[1];
  const float* enc1_w  = (const float*)d_in[2];   // [2][256][128]
  const float* enc1_b  = (const float*)d_in[3];   // [2][128]
  const float* enc2_w  = (const float*)d_in[4];   // [2][128][128]
  const float* enc2_b  = (const float*)d_in[5];   // [2][128]
  const float* mu_w    = (const float*)d_in[6];   // [128][128]
  const float* mu_b    = (const float*)d_in[7];
  const float* zita_w  = (const float*)d_in[8];
  const float* zita_b  = (const float*)d_in[9];
  const float* prelu_w = (const float*)d_in[10];
  const float* agcn1_w = (const float*)d_in[11];  // [2][128][32]
  const float* agcn2_w = (const float*)d_in[12];  // [2][32][10]

  const int n = in_sizes[0] / 256;   // 100000
  const int m = 1024, K = 10;
  const int ntiles = n / 16;         // 6250

  // outputs (flat concat, reference return order)
  float* out    = (float*)d_out;
  size_t zlen   = (size_t)n * 128;
  float* Z0     = out;
  float* Z1     = Z0 + zlen;
  float* anchor = Z1 + zlen;
  float* fuseZ  = anchor + (size_t)m * 128;
  float* A0     = fuseZ + zlen;
  float* A1     = A0 + (size_t)n * m;
  float* Q0     = A1 + (size_t)n * m;
  float* Q1     = Q0 + (size_t)m * K;

  // workspace carve (~10 MB)
  char* wsp = (char*)d_ws;
  auto carve = [&](size_t bytes) {
    char* p = wsp; wsp += (bytes + 255) & ~(size_t)255; return p;
  };
  bf16*  anchorPk  = (bf16*)carve((size_t)m * 128 * 2);       // packed anchor^T
  bf16*  enc1_wp   = (bf16*)carve((size_t)2 * 256 * 128 * 2);
  bf16*  enc2_wp   = (bf16*)carve((size_t)2 * 128 * 128 * 2);
  bf16*  mu_wp     = (bf16*)carve((size_t)128 * 128 * 2);
  bf16*  zita_wp   = (bf16*)carve((size_t)128 * 128 * 2);
  bf16*  agcn1_wp  = (bf16*)carve((size_t)2 * 128 * 32 * 2);
  bf16*  agcn2_wp  = (bf16*)carve((size_t)2 * 32 * 16 * 2);
  int*   top_idx   = (int*)carve((size_t)n * 10 * 4);
  float* top_w     = (float*)carve((size_t)n * 10 * 4);
  float* dcol      = (float*)carve((size_t)m * 4);
  float* dinv      = (float*)carve((size_t)m * 4);
  float* t2        = (float*)carve((size_t)m * 128 * 4);
  float* conv1     = (float*)carve((size_t)m * 128 * 4);
  float* qbuf      = (float*)carve((size_t)m * 32 * 4);
  float* t2b       = (float*)carve((size_t)m * 32 * 4);
  float* conv2     = (float*)carve((size_t)m * 32 * 4);

  // 1) pack weights into WMMA B-fragment layout (bf16)
  auto pack = [&](const float* s, bf16* d, int Kd, int Np, int Nr, int tr) {
    int tot = Kd * Np;
    pack_b_kernel<<<(tot + 255) / 256, 256, 0, stream>>>(s, d, Kd, Np, Nr, tr);
  };
  for (int v = 0; v < 2; ++v) {
    pack(enc1_w + (size_t)v * 256 * 128, enc1_wp + (size_t)v * 256 * 128, 256, 128, 128, 0);
    pack(enc2_w + (size_t)v * 128 * 128, enc2_wp + (size_t)v * 128 * 128, 128, 128, 128, 0);
    pack(agcn1_w + (size_t)v * 128 * 32, agcn1_wp + (size_t)v * 128 * 32, 128, 32, 32, 0);
    pack(agcn2_w + (size_t)v * 32 * 10, agcn2_wp + (size_t)v * 32 * 16, 32, 16, 10, 0);
  }
  pack(mu_w, mu_wp, 128, 128, 128, 0);
  pack(zita_w, zita_wp, 128, 128, 128, 0);

  // 2) encoders
  int eblocks = (ntiles + 7) / 8;
  encoder_kernel<<<eblocks, 256, 0, stream>>>(X0, enc1_wp, enc1_b,
                                              enc2_wp, enc2_b, Z0, ntiles);
  encoder_kernel<<<eblocks, 256, 0, stream>>>(X1, enc1_wp + 256 * 128, enc1_b + 128,
                                              enc2_wp + 128 * 128, enc2_b + 128,
                                              Z1, ntiles);
  // 3) fuse
  fuse_kernel<<<n, 128, 0, stream>>>(Z0, Z1, fuseZ);
  // 4) anchors, then pack anchor^T (B = anchor^T : [128 x 1024], transpose read)
  anchor_kernel<<<8, 256, 0, stream>>>(fuseZ, mu_wp, mu_b, zita_wp, zita_b,
                                       prelu_w, anchor);
  pack(anchor, anchorPk, 128, 1024, 1024, 1);

  // 5) zero both dense A matrices (contiguous region)
  zero4_kernel<<<4096, 256, 0, stream>>>((float4*)A0, (long long)2 * n * m / 4);

  for (int v = 0; v < 2; ++v) {
    const float* Zv = v ? Z1 : Z0;
    float* Av = v ? A1 : A0;
    float* Qv = v ? Q1 : Q0;
    const bf16* w1 = agcn1_wp + (size_t)v * 128 * 32;
    const bf16* w2 = agcn2_wp + (size_t)v * 32 * 16;

    // scores + top-10 + softmax + dense scatter + compact (idx, w)
    score_topk_kernel<<<ntiles, 32, 0, stream>>>(Zv, anchorPk, Av, top_idx, top_w);

    // d = colsum(A); dinv
    zero4_kernel<<<1, 256, 0, stream>>>((float4*)dcol, m / 4);
    colsum_kernel<<<(n * 10 + 255) / 256, 256, 0, stream>>>(top_idx, top_w, dcol,
                                                            n * 10);
    dinv_kernel<<<4, 256, 0, stream>>>(dcol, dinv);

    // conv1 = dinv * A^T (A anchor)
    zero4_kernel<<<128, 256, 0, stream>>>((float4*)t2, (long long)m * 128 / 4);
    conv_scatter_kernel<<<n, 128, 0, stream>>>(top_idx, top_w, anchor, t2, 128);
    scale_kernel<<<(m * 128 + 255) / 256, 256, 0, stream>>>(t2, dinv, conv1, 128,
                                                            m * 128);
    // q = lrelu(conv1 @ agcn1_w)
    agcn1_kernel<<<8, 256, 0, stream>>>(conv1, w1, qbuf);

    // conv2 = dinv * A^T (A q)
    zero4_kernel<<<32, 256, 0, stream>>>((float4*)t2b, (long long)m * 32 / 4);
    conv_scatter_kernel<<<n, 32, 0, stream>>>(top_idx, top_w, qbuf, t2b, 32);
    scale_kernel<<<(m * 32 + 255) / 256, 256, 0, stream>>>(t2b, dinv, conv2, 32,
                                                           m * 32);
    // Q = softmax(conv2 @ agcn2_w)
    agcn2_kernel<<<8, 256, 0, stream>>>(conv2, w2, Qv);
  }
}
